// MultiHeadAttentionLinear_27066883899621
// MI455X (gfx1250) — compile-verified, matrix operand order checked by reading, not run on hardware
//
#include <hip/hip_runtime.h>
#include <hip/hip_bf16.h>

// MI455X / gfx1250, wave32. All GEMMs via v_wmma_f32_16x16x32_bf16 with
// vectorized (b128) fragment loads; f32 accumulate. Attention kernel stages
// its Q tile via CDNA5 async global->LDS copies (ASYNCcnt).

typedef __attribute__((ext_vector_type(16))) __bf16 v16bf;
typedef __attribute__((ext_vector_type(8)))  __bf16 v8bf;
typedef __attribute__((ext_vector_type(8)))  float  v8f;

#define S_LEN 1024
#define D_DIM 512
#define H_NUM 8
#define B_NUM 2
#define NEG_INF -1.0e30f

__device__ __forceinline__ v8f wmma_bf16(v16bf a, v16bf b, v8f c) {
    return __builtin_amdgcn_wmma_f32_16x16x32_bf16(false, a, false, b, (short)0, c,
                                                   false, false);
}

// Load one bf16 16x32 A/B fragment for this lane when the per-lane data is
// K-contiguous (row-major with K innermost). ISA 7.12.2 layout:
//   lanes 0-15 : K = 0..7 (elems 0..7), K = 16..23 (elems 8..15)
//   lanes 16-31: K = 8..15,              K = 24..31
__device__ __forceinline__ v16bf load_frag(const __bf16* row, int koff) {
    v8bf lo = *(const v8bf*)(row + koff);
    v8bf hi = *(const v8bf*)(row + 16 + koff);
    return __builtin_shufflevector(lo, hi, 0, 1, 2, 3, 4, 5, 6, 7,
                                   8, 9, 10, 11, 12, 13, 14, 15);
}

// ---------------------------------------------------------------------------
// Kernel 0: precision/layout prep.
//   z=0: xb  = bf16(x)                     (B*S*D)
//   z=1: yb  = bf16(y)                     (B*S*D)
//   z=2..4: Wqt/Wkt/Wvt[n*K+k] = bf16(W[k*N+n])   (N=4096, K=512)
//   z=5: Wpt[n*K+k] = bf16(Wp[k*N+n])             (N=512,  K=512)
// ---------------------------------------------------------------------------
__global__ void convert_kernel(const float* __restrict__ x, const float* __restrict__ y,
                               const float* __restrict__ Wq, const float* __restrict__ Wk,
                               const float* __restrict__ Wv, const float* __restrict__ Wp,
                               __bf16* __restrict__ xb, __bf16* __restrict__ yb,
                               __bf16* __restrict__ Wqt, __bf16* __restrict__ Wkt,
                               __bf16* __restrict__ Wvt, __bf16* __restrict__ Wpt) {
    const int which = blockIdx.z;
    const size_t idx = (size_t)blockIdx.x * blockDim.x + threadIdx.x;
    if (which < 2) {
        const size_t total = (size_t)B_NUM * S_LEN * D_DIM;
        if (idx >= total) return;
        const float* src = which ? y : x;
        __bf16* dst = which ? yb : xb;
        dst[idx] = (__bf16)src[idx];
    } else {
        const int N = (which == 5) ? D_DIM : (H_NUM * D_DIM);
        const size_t total = (size_t)N * D_DIM;
        if (idx >= total) return;
        const float* src = (which == 2) ? Wq : (which == 3) ? Wk
                         : (which == 4) ? Wv : Wp;
        __bf16* dst = (which == 2) ? Wqt : (which == 3) ? Wkt
                    : (which == 4) ? Wvt : Wpt;
        const int n = (int)(idx >> 9);          // K = 512
        const int k = (int)(idx & (D_DIM - 1));
        dst[idx] = (__bf16)src[(size_t)k * N + n];
    }
}

// ---------------------------------------------------------------------------
// Kernel 1: fused projections, 32x32 output per wave (4 wmma / K-step).
//   z=0: q -> qb (h,b,s,d) bf16      z=1: k -> kb (h,b,s,d) bf16
//   z=2: v -> vT ((h,b),d,s) bf16    z=3: xp -> (b,s,d) f32
// ---------------------------------------------------------------------------
__global__ void proj_kernel(const __bf16* __restrict__ xb, const __bf16* __restrict__ yb,
                            const __bf16* __restrict__ Wqt, const __bf16* __restrict__ Wkt,
                            const __bf16* __restrict__ Wvt, const __bf16* __restrict__ Wpt,
                            const float* __restrict__ bq, const float* __restrict__ bk,
                            const float* __restrict__ bv,
                            __bf16* __restrict__ qb, __bf16* __restrict__ kb,
                            __bf16* __restrict__ vT, float* __restrict__ xp) {
    const int which = blockIdx.z;
    const int N = (which == 3) ? D_DIM : (H_NUM * D_DIM);
    if (blockIdx.x * 32 >= N) return;

    const __bf16* A  = (which == 1 || which == 2) ? yb : xb;
    const __bf16* Wt = (which == 0) ? Wqt : (which == 1) ? Wkt
                     : (which == 2) ? Wvt : Wpt;
    const float* bias = (which == 0) ? bq : (which == 1) ? bk
                      : (which == 2) ? bv : nullptr;

    const int lane = threadIdx.x;
    const int m0 = blockIdx.y * 32;
    const int n0 = blockIdx.x * 32;
    const int l15 = lane & 15;
    const int koff = (lane >> 4) << 3;

    const __bf16* arow0 = A  + (size_t)(m0 + l15) * D_DIM;
    const __bf16* arow1 = arow0 + (size_t)16 * D_DIM;
    const __bf16* brow0 = Wt + (size_t)(n0 + l15) * D_DIM;
    const __bf16* brow1 = brow0 + (size_t)16 * D_DIM;

    v8f acc00 = {}, acc01 = {}, acc10 = {}, acc11 = {};
#pragma unroll 4
    for (int k0 = 0; k0 < D_DIM; k0 += 32) {
        if ((k0 & 127) == 0 && k0 + 128 < D_DIM) {
            __builtin_prefetch(arow0 + k0 + 128, 0, 0);  // global_prefetch_b8
            __builtin_prefetch(brow0 + k0 + 128, 0, 0);
        }
        v16bf a0 = load_frag(arow0 + k0, koff);
        v16bf a1 = load_frag(arow1 + k0, koff);
        v16bf b0 = load_frag(brow0 + k0, koff);
        v16bf b1 = load_frag(brow1 + k0, koff);
        acc00 = wmma_bf16(a0, b0, acc00);
        acc01 = wmma_bf16(a0, b1, acc01);
        acc10 = wmma_bf16(a1, b0, acc10);
        acc11 = wmma_bf16(a1, b1, acc11);
    }

    const int mbase = (lane >> 4) << 3;  // C/D layout: lanes 0-15 -> M r, 16-31 -> M 8+r
#pragma unroll
    for (int ti = 0; ti < 2; ++ti) {
#pragma unroll
        for (int tj = 0; tj < 2; ++tj) {
            v8f c = (ti == 0) ? (tj == 0 ? acc00 : acc01)
                              : (tj == 0 ? acc10 : acc11);
            const int n = n0 + tj * 16 + l15;
            const float bs = bias ? bias[n] : 0.0f;
            const int mg = m0 + ti * 16 + mbase;   // first of 8 consecutive rows
            const int b_ = mg >> 10;
            const int s_ = mg & (S_LEN - 1);
            if (which == 3) {
#pragma unroll
                for (int r = 0; r < 8; ++r)
                    xp[(size_t)(mg + r) * D_DIM + n] = c[r] + bs;
            } else if (which == 2) {
                // vT[((h*B+b)*D + d) * S + s], s contiguous over r -> one b128
                const int h = n >> 9, d = n & (D_DIM - 1);
                v8bf pk;
#pragma unroll
                for (int r = 0; r < 8; ++r) pk[r] = (__bf16)(c[r] + bs);
                *(v8bf*)(vT + ((size_t)(h * B_NUM + b_) * D_DIM + d) * S_LEN + s_) = pk;
            } else {
                const int h = n >> 9, d = n & (D_DIM - 1);
                __bf16* dst = (which == 0) ? qb : kb;
                const size_t base =
                    (((size_t)(h * B_NUM + b_) * S_LEN) + s_) * D_DIM + d;
#pragma unroll
                for (int r = 0; r < 8; ++r)
                    dst[base + (size_t)r * D_DIM] = (__bf16)(c[r] + bs);
            }
        }
    }
}

// ---------------------------------------------------------------------------
// Kernel 2: energy = q @ k^T per (h,b), 16 rows per CTA (8 waves).
// The 16x512 bf16 Q tile (16 KB) is staged into LDS ONCE per block using
// CDNA5 async global->LDS copies (global_load_async_to_lds_b128, ASYNCcnt),
// then every wave reads its A-fragments from LDS (ds_load_b128 pairs) while
// streaming K fragments from global. Energy (16x1024 f32, 64 KB LDS) gets a
// masked row-softmax via wave32 shuffles; attn written TRANSPOSED [t][s].
// Dynamic LDS: 64 KB energy + 16 KB Q = 80 KB (WGP has 320 KB).
// ---------------------------------------------------------------------------
__global__ void attn_kernel(const __bf16* __restrict__ qb,
                            const __bf16* __restrict__ kb,
                            const int* __restrict__ mask,
                            __bf16* __restrict__ attnT) {
    extern __shared__ char smem[];
    float*  se = (float*)smem;                       // 16 x 1024 f32 (64 KB)
    __bf16* qs = (__bf16*)(smem + 16 * S_LEN * 4);   // 16 x 512 bf16 (16 KB)

    const int hb = blockIdx.y;
    const int s0 = blockIdx.x * 16;
    const int tid = threadIdx.x;
    const int wave = tid >> 5;
    const int lane = tid & 31;
    const int l15 = lane & 15;
    const int koff = (lane >> 4) << 3;
    const int mbase = koff;

    const __bf16* Q = qb + (size_t)hb * S_LEN * D_DIM;  // [s][d]
    const __bf16* K = kb + (size_t)hb * S_LEN * D_DIM;  // [t][d]

    // ---- Async-stage the Q row block into LDS (1024 x 16-byte chunks). ----
    {
        const __bf16* src = Q + (size_t)s0 * D_DIM;   // 16384 contiguous bytes
        const unsigned qs_base = (unsigned)(uintptr_t)qs;  // low 32 bits = LDS addr
        for (int i = tid; i < 1024; i += 256) {
            const unsigned ldsa = qs_base + (unsigned)i * 16u;
            const __bf16* g = src + (size_t)i * 8;
            asm volatile("global_load_async_to_lds_b128 %0, %1, off"
                         :: "v"(ldsa), "v"(g)
                         : "memory");
        }
        asm volatile("s_wait_asynccnt 0" ::: "memory");
    }
    __syncthreads();

    const __bf16* arow = qs + (size_t)l15 * D_DIM;      // A row in LDS

    for (int tsup = wave; tsup < S_LEN / 32; tsup += 8) {
        const int n0 = tsup * 32;
        const __bf16* brow0 = K + (size_t)(n0 + l15) * D_DIM;
        const __bf16* brow1 = brow0 + (size_t)16 * D_DIM;
        v8f c0 = {}, c1 = {};
#pragma unroll 4
        for (int k0 = 0; k0 < D_DIM; k0 += 32) {
            v16bf a  = load_frag(arow + k0, koff);      // ds_load_b128 x2
            v16bf b0 = load_frag(brow0 + k0, koff);
            v16bf b1 = load_frag(brow1 + k0, koff);
            c0 = wmma_bf16(a, b0, c0);
            c1 = wmma_bf16(a, b1, c1);
        }
#pragma unroll
        for (int r = 0; r < 8; ++r) {
            se[(mbase + r) * S_LEN + n0 + l15]      = c0[r];
            se[(mbase + r) * S_LEN + n0 + 16 + l15] = c1[r];
        }
    }
    __syncthreads();

    // Softmax: 8 waves x 2 rows each.
#pragma unroll
    for (int rr = 0; rr < 2; ++rr) {
        const int r = wave * 2 + rr;
        float* row = &se[r * S_LEN];

        float mx = -3.0e38f;
        for (int t = lane; t < S_LEN; t += 32) {
            float e = mask[t] ? row[t] : NEG_INF;
            row[t] = e;
            mx = fmaxf(mx, e);
        }
#pragma unroll
        for (int off = 16; off > 0; off >>= 1)
            mx = fmaxf(mx, __shfl_xor(mx, off, 32));

        float sum = 0.0f;
        for (int t = lane; t < S_LEN; t += 32)
            sum += __expf(row[t] - mx);
#pragma unroll
        for (int off = 16; off > 0; off >>= 1)
            sum += __shfl_xor(sum, off, 32);

        const float inv = 1.0f / sum;
        __bf16* dst = attnT + (size_t)hb * S_LEN * S_LEN + (s0 + r);
        for (int t = lane; t < S_LEN; t += 32)
            dst[(size_t)t * S_LEN] = (__bf16)(__expf(row[t] - mx) * inv);
    }
}

// ---------------------------------------------------------------------------
// Kernel 3: out = attn^T @ v then (gamma*out + xp)/(gamma+1).
// attnT is [t][s] and vT is [c][s], so both fragments are K-contiguous b128
// loads. 32x32 per wave (4 wmma / K-step), K = 1024. Output (B,H,S,D) f32.
// ---------------------------------------------------------------------------
__global__ void out_kernel(const __bf16* __restrict__ attnT,
                           const __bf16* __restrict__ vT,
                           const float* __restrict__ xp,
                           const float* __restrict__ gamma,
                           float* __restrict__ out) {
    const int hb = blockIdx.z;
    const int h = hb >> 1;
    const int b = hb & 1;
    const int t0 = blockIdx.y * 32;
    const int c0 = blockIdx.x * 32;
    const int lane = threadIdx.x;
    const int l15 = lane & 15;
    const int koff = (lane >> 4) << 3;

    const __bf16* A = attnT + (size_t)hb * S_LEN * S_LEN;  // [t][s]
    const __bf16* V = vT + (size_t)hb * D_DIM * S_LEN;     // [c][s]
    const __bf16* arow0 = A + (size_t)(t0 + l15) * S_LEN;
    const __bf16* arow1 = arow0 + (size_t)16 * S_LEN;
    const __bf16* brow0 = V + (size_t)(c0 + l15) * S_LEN;
    const __bf16* brow1 = brow0 + (size_t)16 * S_LEN;

    v8f acc00 = {}, acc01 = {}, acc10 = {}, acc11 = {};
#pragma unroll 4
    for (int k0 = 0; k0 < S_LEN; k0 += 32) {
        if ((k0 & 127) == 0 && k0 + 128 < S_LEN) {
            __builtin_prefetch(arow0 + k0 + 128, 0, 0);  // global_prefetch_b8
            __builtin_prefetch(brow0 + k0 + 128, 0, 0);
        }
        v16bf a0 = load_frag(arow0 + k0, koff);
        v16bf a1 = load_frag(arow1 + k0, koff);
        v16bf b0 = load_frag(brow0 + k0, koff);
        v16bf b1 = load_frag(brow1 + k0, koff);
        acc00 = wmma_bf16(a0, b0, acc00);
        acc01 = wmma_bf16(a0, b1, acc01);
        acc10 = wmma_bf16(a1, b0, acc10);
        acc11 = wmma_bf16(a1, b1, acc11);
    }

    const float g = gamma[h];
    const float rg = 1.0f / (g + 1.0f);
    const int mbase = (lane >> 4) << 3;
#pragma unroll
    for (int ti = 0; ti < 2; ++ti) {
#pragma unroll
        for (int tj = 0; tj < 2; ++tj) {
            v8f c = (ti == 0) ? (tj == 0 ? acc00 : acc01)
                              : (tj == 0 ? acc10 : acc11);
            const int cc = c0 + tj * 16 + l15;
            const int tg = t0 + ti * 16 + mbase;
#pragma unroll
            for (int r = 0; r < 8; ++r) {
                const int t = tg + r;
                const float o =
                    (g * c[r] + xp[((size_t)b * S_LEN + t) * D_DIM + cc]) * rg;
                out[(((size_t)(b * H_NUM + h) * S_LEN) + t) * D_DIM + cc] = o;
            }
        }
    }
}

// ---------------------------------------------------------------------------
// Workspace layout (~100.5 MB): xb,yb | Wqt,Wkt,Wvt,Wpt | qb,kb,vT | attnT | xp
// ---------------------------------------------------------------------------
extern "C" void kernel_launch(void* const* d_in, const int* in_sizes, int n_in,
                              void* d_out, int out_size, void* d_ws, size_t ws_size,
                              hipStream_t stream) {
    const float* x     = (const float*)d_in[0];
    const float* y     = (const float*)d_in[1];
    const float* Wq    = (const float*)d_in[2];
    const float* bq    = (const float*)d_in[3];
    const float* Wk    = (const float*)d_in[4];
    const float* bk    = (const float*)d_in[5];
    const float* Wv    = (const float*)d_in[6];
    const float* bv    = (const float*)d_in[7];
    const float* Wp    = (const float*)d_in[8];
    const float* gamma = (const float*)d_in[9];
    const int*   mask  = (const int*)d_in[10];

    char* ws = (char*)d_ws;
    const size_t XY = (size_t)B_NUM * S_LEN * D_DIM;          // 1,048,576
    const size_t WN = (size_t)H_NUM * D_DIM * D_DIM;          // 2,097,152
    const size_t QK = (size_t)H_NUM * B_NUM * S_LEN * D_DIM;  // 8,388,608
    size_t off = 0;
    __bf16* xb   = (__bf16*)(ws + off); off += XY * 2;
    __bf16* yb   = (__bf16*)(ws + off); off += XY * 2;
    __bf16* Wqt  = (__bf16*)(ws + off); off += WN * 2;
    __bf16* Wkt  = (__bf16*)(ws + off); off += WN * 2;
    __bf16* Wvt  = (__bf16*)(ws + off); off += WN * 2;
    __bf16* Wpt  = (__bf16*)(ws + off); off += (size_t)D_DIM * D_DIM * 2;
    __bf16* qb   = (__bf16*)(ws + off); off += QK * 2;
    __bf16* kb   = (__bf16*)(ws + off); off += QK * 2;
    __bf16* vT   = (__bf16*)(ws + off); off += QK * 2;
    __bf16* attnT= (__bf16*)(ws + off); off += (size_t)H_NUM * B_NUM * S_LEN * S_LEN * 2;
    float*  xp   = (float*)(ws + off);
    float* out = (float*)d_out;

    // Prep: convert + transpose. Largest plane = 2,097,152 elems -> 8192 blocks.
    convert_kernel<<<dim3(8192, 1, 6), 256, 0, stream>>>(x, y, Wq, Wk, Wv, Wp,
                                                         xb, yb, Wqt, Wkt, Wvt, Wpt);

    // Projections: 32x32 tiles; z=3 (xp) uses only 16 n-supertiles (guarded).
    proj_kernel<<<dim3(128, 64, 4), 32, 0, stream>>>(xb, yb, Wqt, Wkt, Wvt, Wpt,
                                                     bq, bk, bv, qb, kb, vT, xp);

    // Attention + masked softmax: 80 KB dynamic LDS (64 KB energy + 16 KB Q).
    const size_t attn_lds = (size_t)16 * S_LEN * 4 + (size_t)16 * D_DIM * 2;
    attn_kernel<<<dim3(S_LEN / 16, H_NUM * B_NUM), 256, attn_lds, stream>>>(
        qb, kb, mask, attnT);

    // Output GEMM + gamma blend.
    out_kernel<<<dim3(D_DIM / 32, S_LEN / 32, H_NUM * B_NUM), 32, 0, stream>>>(
        attnT, vT, xp, gamma, out);
}
